// TwoBodyPadAttn_52295521796555
// MI455X (gfx1250) — matrix operand order repaired and checked by reading, not compiled, not run on hardware
//
#include <hip/hip_runtime.h>

// ---------------------------------------------------------------------------
// TwoBodyPadAttn for MI455X (gfx1250, wave32, WMMA)
// B=16, N=512, C=256, H=8, D=32.  Softmax is over the BATCH axis (axis=0).
// Pipeline: prep (bf16 transpose/convert, zero-padded weights) ->
//           proj GEMM (WMMA, branchless inner loop) ->
//           fused batch-softmax attention (WMMA) -> LayerNorm.
// ---------------------------------------------------------------------------

typedef __bf16 bf16_t;
typedef __attribute__((ext_vector_type(16))) __bf16 v16bf;
typedef __attribute__((ext_vector_type(8)))  float  v8f;

#define DEV __device__ __forceinline__

static constexpr int Bb = 16, Nn = 512, Cc = 256, Hh = 8, Dd = 32;
static constexpr int NOUT = 784;  // 256 (Q) + 512 (KV) + 16 (EG)
static constexpr int NPAD = 832;  // padded to 13 groups x 64 cols

// ---- WMMA fragment loaders (ISA 7.12.2 layouts, wave32) -------------------
// A (16x32 bf16): lane m=lane&15, kh=lane>>4; elems 0..7 = K 8*kh..+7,
//                 elems 8..15 = K 16+8*kh..+7  -> two b128 loads.
DEV v16bf load_a_frag(const bf16_t* base, int row_stride, int lane) {
  const int m = lane & 15, kh = lane >> 4;
  const bf16_t* p = base + (size_t)m * row_stride;
  union { v16bf v; uint4 u[2]; } r;
  r.u[0] = *reinterpret_cast<const uint4*>(p + 8 * kh);
  r.u[1] = *reinterpret_cast<const uint4*>(p + 16 + 8 * kh);
  return r.v;
}
// B (32x16 bf16): lane n=lane&15, kh=lane>>4; elems 0..7 = K 8*kh..+7,
//                 elems 8..15 = K 16+8*kh..+7 from [n][k] -> two b128 loads.
DEV v16bf load_b_frag(const bf16_t* base, int row_stride, int lane) {
  const int n = lane & 15, kh = lane >> 4;
  const bf16_t* p = base + (size_t)n * row_stride + 8 * kh;
  union { v16bf v; uint4 u[2]; } r;
  r.u[0] = *reinterpret_cast<const uint4*>(p);
  r.u[1] = *reinterpret_cast<const uint4*>(p + 16);
  return r.v;
}

DEV v8f wmma_bf16(v16bf a, v16bf b, v8f c) {
  return __builtin_amdgcn_wmma_f32_16x16x32_bf16(false, a, false, b,
                                                 (short)0, c, false, false);
}

// ---------------------------------------------------------------------------
// Kernel 0: prep.  blockIdx.x < NPAD: build Wt[n][k] (bf16, k contiguous,
// transposed from Wq/Wkv/Weg, zero for pad cols) and biasAll[n].
// Else: featbf[row][c] = bf16(feat).  One-time ~9 MB of traffic.
// ---------------------------------------------------------------------------
__global__ __launch_bounds__(256) void prep_kernel(
    const float* __restrict__ feat,
    const float* __restrict__ Wq,  const float* __restrict__ bq,
    const float* __restrict__ Wkv, const float* __restrict__ bkv,
    const float* __restrict__ Weg, const float* __restrict__ beg,
    bf16_t* __restrict__ featbf, bf16_t* __restrict__ Wt,
    float* __restrict__ biasAll)
{
  const int bid = blockIdx.x, k = threadIdx.x;
  if (bid < NPAD) {
    const int n = bid;
    float w = 0.f, bv = 0.f;
    if (n < 256)      { w = Wq[(size_t)k * 256 + n];          bv = bq[n];        }
    else if (n < 768) { w = Wkv[(size_t)k * 512 + (n - 256)]; bv = bkv[n - 256]; }
    else if (n < NOUT){ w = Weg[(size_t)k * 16 + (n - 768)];  bv = beg[n - 768]; }
    Wt[(size_t)n * Cc + k] = (bf16_t)w;
    if (k == 0) biasAll[n] = bv;
  } else {
    const int row = bid - NPAD;
    featbf[(size_t)row * Cc + k] = (bf16_t)feat[(size_t)row * Cc + k];
  }
}

// ---------------------------------------------------------------------------
// Kernel 1: projection GEMM  Y = feat @ [Wq | Wkv | Weg] + bias
// M=8192, K=256, Nout=784 (padded 832).  Each wave: one 16x64 tile, the
// k-loop is branchless: 1 A-frag (b128 x2) + 4 B-frags + 4 WMMAs per step.
// Epilogue scatters into head-major bf16 layouts + fp32 E / sigmoid(G).
// ---------------------------------------------------------------------------
__global__ __launch_bounds__(128) void proj_kernel(
    const bf16_t* __restrict__ featbf, const bf16_t* __restrict__ Wt,
    const float* __restrict__ biasAll,
    bf16_t* __restrict__ Qbf, bf16_t* __restrict__ Kbf, bf16_t* __restrict__ Vt,
    float* __restrict__ Eo, float* __restrict__ Go)
{
  const int wave = threadIdx.x >> 5, lane = threadIdx.x & 31;
  const int ln16 = lane & 15, kh = lane >> 4;
  const int m0 = (blockIdx.x * 4 + wave) * 16;   // 512 m-tiles via 128x4 waves
  const int n0 = blockIdx.y * 64;                // 13 n-groups of 64 cols

  v8f acc[4] = {{}, {}, {}, {}};
  for (int k0 = 0; k0 < Cc; k0 += 32) {
    const v16bf a = load_a_frag(featbf + (size_t)m0 * Cc + k0, Cc, lane);
#pragma unroll
    for (int t = 0; t < 4; ++t) {
      const v16bf wb = load_b_frag(Wt + (size_t)(n0 + 16 * t) * Cc + k0, Cc, lane);
      acc[t] = wmma_bf16(a, wb, acc[t]);
    }
  }

  constexpr float qscale = 0.17677669529663687f; // D^-0.5
#pragma unroll
  for (int t = 0; t < 4; ++t) {
    if (n0 + 16 * t >= NOUT) continue;           // pad tiles: discard
    const int col = n0 + 16 * t + ln16;          // region-uniform per tile
    const float bias = biasAll[col];
#pragma unroll
    for (int r = 0; r < 8; ++r) {
      const int row = m0 + r + 8 * kh;           // token index 0..8191
      const int b = row >> 9, i = row & (Nn - 1);
      const float y = acc[t][r] + bias;
      if (col < 256) {                           // Q: [b][h][i][d], pre-scaled
        const int h = col & 7, d = col >> 3;
        Qbf[(((size_t)(b * Hh + h) * Nn) + i) * Dd + d] = (bf16_t)(y * qscale);
      } else if (col < 512) {                    // K: [b][h][j][d]
        const int c2 = col - 256, h = c2 & 7, d = c2 >> 3;
        Kbf[(((size_t)(b * Hh + h) * Nn) + i) * Dd + d] = (bf16_t)y;
      } else if (col < 768) {                    // V transposed: [b][h][d][j]
        const int c2 = col - 512, h = c2 & 7, d = c2 >> 3;
        Vt[(((size_t)(b * Hh + h) * Dd) + d) * Nn + i] = (bf16_t)y;
      } else if (col < 776) {                    // E: [b][h][j] fp32
        Eo[((size_t)(b * Hh + (col - 768))) * Nn + i] = y;
      } else {                                   // sigmoid(G): [b][h][j] fp32
        Go[((size_t)(b * Hh + (col - 776))) * Nn + i] =
            1.f / (1.f + __expf(-y));
      }
    }
  }
}

// ---------------------------------------------------------------------------
// Kernel 2: fused attention with softmax over the batch axis.
// Block = 16 waves = 16 batches, one (i-tile of 16, head h) per block.
// Per 32-wide j step: QK^T WMMA -> +E, mask -> LDS -> cross-batch max/sumexp
// -> A = exp/sum * sigG -> bf16 LDS -> A*V WMMA accumulate.
// Avoids ever materializing the 128 MB logits tensor in HBM.
// ---------------------------------------------------------------------------
__global__ __launch_bounds__(512) void attn_kernel(
    const bf16_t* __restrict__ Qbf, const bf16_t* __restrict__ Kbf,
    const bf16_t* __restrict__ Vt,  const float* __restrict__ Eb,
    const float* __restrict__ sGb,  const unsigned char* __restrict__ mask,
    float* __restrict__ Va)
{
  __shared__ float  lds_S[16][16][33];          // [b][i][j] logits (+pad)
  __shared__ float  lds_ms[2][16][32];          // max / sumexp over b
  __shared__ __align__(16) bf16_t lds_A[16][16][32];  // [b][i][j] probs

  const int i0   = blockIdx.x * 16;
  const int h    = blockIdx.y;
  const int b    = threadIdx.x >> 5;            // wave == batch
  const int lane = threadIdx.x & 31;
  const int ln16 = lane & 15, kh = lane >> 4;
  const float NEG_INF = -__builtin_inff();

  const bf16_t* qbase = Qbf + ((size_t)(b * Hh + h) * Nn + i0) * Dd;
  const bf16_t* kbase = Kbf + ((size_t)(b * Hh + h) * Nn) * Dd;
  const bf16_t* vbase = Vt  + ((size_t)(b * Hh + h) * Dd) * Nn;
  const float*  ebase = Eb  + (size_t)(b * Hh + h) * Nn;
  const float*  gbase = sGb + (size_t)(b * Hh + h) * Nn;

  const v16bf aq = load_a_frag(qbase, Dd, lane);   // resident all iterations
  v8f acc0 = {}, acc1 = {};

  for (int j0 = 0; j0 < Nn; j0 += 32) {
    v8f s[2];
#pragma unroll
    for (int half = 0; half < 2; ++half) {
      const int jb = j0 + 16 * half;
      const v16bf bk = load_b_frag(kbase + (size_t)jb * Dd, Dd, lane);
      v8f z = {};
      v8f sv = wmma_bf16(aq, bk, z);               // S[i][j] = Q·K over d
      const int j = jb + ln16;
      const float e = ebase[j];                    // E broadcast over i
      const unsigned char* mrow = mask + ((size_t)b * Nn + i0) * Nn + j;
#pragma unroll
      for (int r = 0; r < 8; ++r) {
        const int irow = r + 8 * kh;
        const float val = sv[r] + e;
        sv[r] = mrow[(size_t)irow * Nn] ? val : NEG_INF;
        lds_S[b][irow][16 * half + ln16] = sv[r];
      }
      s[half] = sv;
    }
    __syncthreads();

    // Softmax stats across the 16 batches: 512 threads = 16i x 32j pairs.
    {
      const int i  = threadIdx.x >> 5;
      const int jl = threadIdx.x & 31;
      float vbuf[16], mx = NEG_INF;
#pragma unroll
      for (int bb = 0; bb < 16; ++bb) {
        vbuf[bb] = lds_S[bb][i][jl];
        mx = fmaxf(mx, vbuf[bb]);
      }
      float sum = 0.f;
#pragma unroll
      for (int bb = 0; bb < 16; ++bb) sum += __expf(vbuf[bb] - mx);
      lds_ms[0][i][jl] = mx;
      lds_ms[1][i][jl] = sum;
    }
    __syncthreads();

    // A[b][i][j] = exp(s - m)/sum * sigmoid(G[b,j])  -> bf16 A-fragment tile
#pragma unroll
    for (int half = 0; half < 2; ++half) {
      const int jl = 16 * half + ln16;
      const float g = gbase[j0 + jl];
      const v8f sv = s[half];
#pragma unroll
      for (int r = 0; r < 8; ++r) {
        const int irow = r + 8 * kh;
        const float m = lds_ms[0][irow][jl];
        const float d = lds_ms[1][irow][jl];
        lds_A[b][irow][jl] = (bf16_t)(__expf(sv[r] - m) / d * g);
      }
    }
    __syncthreads();

    // Va[i][d] += A[i][j] * V[j][d]  (K = j = 32 per step; d split 2x16)
    const v16bf ap  = load_a_frag(&lds_A[b][0][0], 32, lane);
    const v16bf bv0 = load_b_frag(vbase + j0, Nn, lane);                   // d 0..15
    const v16bf bv1 = load_b_frag(vbase + (size_t)16 * Nn + j0, Nn, lane); // d 16..31
    acc0 = wmma_bf16(ap, bv0, acc0);
    acc1 = wmma_bf16(ap, bv1, acc1);
    __syncthreads();   // protect lds_S/lds_A for next iteration
  }

  // Scatter Va[b, i, c = d*H + h]
#pragma unroll
  for (int r = 0; r < 8; ++r) {
    const int irow = i0 + r + 8 * kh;
    const size_t rowoff = ((size_t)b * Nn + irow) * Cc;
    Va[rowoff + (ln16)      * Hh + h] = acc0[r];
    Va[rowoff + (16 + ln16) * Hh + h] = acc1[r];
  }
}

// ---------------------------------------------------------------------------
// Kernel 3: LayerNorm over C=256 (population variance), fp32 out.
// ---------------------------------------------------------------------------
__global__ __launch_bounds__(256) void ln_kernel(
    const float* __restrict__ Va, const float* __restrict__ gamma,
    const float* __restrict__ beta, float* __restrict__ out)
{
  const int row = blockIdx.x, c = threadIdx.x;
  const float x = Va[(size_t)row * Cc + c];
  __shared__ float rs[256], rq[256];
  rs[c] = x; rq[c] = x * x;
  __syncthreads();
  for (int off = 128; off > 0; off >>= 1) {
    if (c < off) { rs[c] += rs[c + off]; rq[c] += rq[c + off]; }
    __syncthreads();
  }
  const float mean = rs[0] * (1.f / 256.f);
  const float var  = rq[0] * (1.f / 256.f) - mean * mean;
  out[(size_t)row * Cc + c] =
      (x - mean) * rsqrtf(var + 1e-3f) * gamma[c] + beta[c];
}

// ---------------------------------------------------------------------------
extern "C" void kernel_launch(void* const* d_in, const int* in_sizes, int n_in,
                              void* d_out, int out_size, void* d_ws, size_t ws_size,
                              hipStream_t stream) {
  const float* feat = (const float*)d_in[0];
  const unsigned char* mask = (const unsigned char*)d_in[1];
  const float* Wq  = (const float*)d_in[2];
  const float* bq  = (const float*)d_in[3];
  const float* Wkv = (const float*)d_in[4];
  const float* bkv = (const float*)d_in[5];
  const float* Weg = (const float*)d_in[6];
  const float* beg = (const float*)d_in[7];
  const float* gamma = (const float*)d_in[8];
  const float* beta  = (const float*)d_in[9];
  float* out = (float*)d_out;

  // Workspace layout (bytes):
  //   Qbf    [0,    4MB)      bf16 [b][h][n][d]
  //   Kbf    [4MB,  8MB)      bf16 [b][h][n][d]
  //   Vt     [8MB,  12MB)     bf16 [b][h][d][n]
  //   E      [12MB, +256KB)   fp32 [b][h][n]
  //   sigG   [+256KB)         fp32 [b][h][n]
  //   Va     [12.5MB, +8MB)   fp32 [b][n][C]
  //   featbf [20.5MB, +4MB)   bf16 [token][c]
  //   Wt     [24.5MB, +416KB) bf16 [n][k], 832 rows (48 zero pad)
  //   biasAll[25MB, +3.3KB]   fp32 [832]
  char* ws = (char*)d_ws;
  const size_t MB = 1024 * 1024;
  bf16_t* Qbf    = (bf16_t*)(ws);
  bf16_t* Kbf    = (bf16_t*)(ws + 4 * MB);
  bf16_t* Vt     = (bf16_t*)(ws + 8 * MB);
  float*  Eo     = (float* )(ws + 12 * MB);
  float*  Go     = (float* )(ws + 12 * MB + 256 * 1024);
  float*  Va     = (float* )(ws + 12 * MB + 512 * 1024);
  bf16_t* featbf = (bf16_t*)(ws + 20 * MB + 512 * 1024);
  bf16_t* Wt     = (bf16_t*)(ws + 24 * MB + 512 * 1024);
  float*  biasAll= (float* )(ws + 25 * MB);

  // 0) prep: 832 weight rows (incl. zero pad) + 8192 feat rows
  prep_kernel<<<NPAD + Bb * Nn, 256, 0, stream>>>(feat, Wq, bq, Wkv, bkv,
                                                  Weg, beg, featbf, Wt, biasAll);
  // 1) projections: 512 m-tiles via (128 blocks x 4 waves), 13 n-groups of 64
  proj_kernel<<<dim3(128, 13), 128, 0, stream>>>(featbf, Wt, biasAll,
                                                 Qbf, Kbf, Vt, Eo, Go);
  // 2) attention: 32 i-tiles x 8 heads, 16 waves (one per batch)
  attn_kernel<<<dim3(Nn / 16, Hh), 512, 0, stream>>>(Qbf, Kbf, Vt, Eo, Go,
                                                     mask, Va);
  // 3) layernorm: one block per (b, i) row
  ln_kernel<<<Bb * Nn, 256, 0, stream>>>(Va, gamma, beta, out);
}
